// InnerPatchSoftShiftTriple_52501680227057
// MI455X (gfx1250) — matrix operand order repaired and checked by reading, not compiled, not run on hardware
//
#include <hip/hip_runtime.h>
#include <hip/hip_bf16.h>
#include <stdint.h>

// Problem constants (B=2, c=64, H=W=64)
#define BATCH 2
#define CCH   64
#define LL    4096
#define NCH   18      // 576 / 32 K-chunks for the fused-score GEMM
#define QBLK  256     // LL/16 query blocks
#define KBLK  256     // LL/16 key blocks
#define KPAIR 128     // LL/32 key pairs

typedef __attribute__((ext_vector_type(16))) __bf16 v16bf;
typedef __attribute__((ext_vector_type(8)))  float  v8f;

struct alignas(32) Frag2 { uint4 lo, hi; };

static __device__ __forceinline__ v16bf ld_frag(const uint4* p) {
  Frag2 f; f.lo = p[0]; f.hi = p[1];
  return __builtin_bit_cast(v16bf, f);
}

static __device__ __forceinline__ v8f wmma_bf16(v16bf a, v16bf b, v8f c) {
  return __builtin_amdgcn_wmma_f32_16x16x32_bf16(false, a, false, b, (short)0, c,
                                                 false, false);
}

static __device__ __forceinline__ unsigned short f2bf(float f) {
  unsigned u = __float_as_uint(f);
  u = (u + 0x7FFFu + ((u >> 16) & 1u)) >> 16;   // round-to-nearest-even
  return (unsigned short)u;
}

// fuse-shift index map: first conv shifts row-major flat index by t,
// second conv shifts transposed (col-major) flat index by u.
static __device__ __forceinline__ int gshift(int l, int u, int t, bool& valid) {
  int h = l >> 6, w = l & 63;
  int a = (w << 6) + h + u;                 // Tr(l) + u
  if (a < 0 || a >= LL) { valid = false; return 0; }
  int w2 = a >> 6, h2 = a & 63;             // Tr^-1
  int g = (h2 << 6) + w2 + t;               // + t in row-major flat space
  if (g < 0 || g >= LL) { valid = false; return 0; }
  valid = true; return g;
}

// ---------------- prep kernels ----------------

// inverse column norm of latter: invn[b][l]
__global__ void k_invnorm(const float* __restrict__ x, float* __restrict__ invn) {
  int idx = blockIdx.x * blockDim.x + threadIdx.x;
  if (idx >= BATCH * LL) return;
  int b = idx / LL, l = idx % LL;
  const float* lat = x + ((size_t)b * 2 * CCH + CCH) * LL + l;
  float s = 0.f;
  for (int c = 0; c < CCH; ++c) { float v = lat[(size_t)c * LL]; s += v * v; }
  invn[idx] = 1.0f / fmaxf(sqrtf(s), 1e-4f);
}

// Pack Lat9 query-side A fragments: pa[b][qb][ch][lane][dw] (dwords, 2 bf16 each)
__global__ void k_packA(const float* __restrict__ x, unsigned* __restrict__ pa) {
  int idx = blockIdx.x * blockDim.x + threadIdx.x;
  if (idx >= BATCH * QBLK * NCH * 256) return;
  int dw = idx & 7, lane = (idx >> 3) & 31;
  int fragIdx = idx >> 8;
  int ch = fragIdx % NCH; int rest = fragIdx / NCH;
  int qb = rest % QBLK;  int b = rest / QBLK;
  int m = lane & 15, half = lane >> 4, grp = dw >> 2;
  int kk0 = grp * 16 + half * 8 + (dw & 3) * 2;   // ISA 16-bit A 16x32 layout
  int l = qb * 16 + m;
  unsigned outw = 0;
  for (int e = 0; e < 2; ++e) {
    int j = ch * 32 + kk0 + e;          // j = s*64 + c'
    int s = j >> 6, c = j & 63;
    int u = s / 3 - 1, t = s % 3 - 1;
    bool v; int g = gshift(l, u, t, v);
    float val = v ? x[((size_t)(b * 2 * CCH + CCH + c)) * LL + g] : 0.f;
    outw |= (unsigned)f2bf(val) << (16 * e);
  }
  pa[idx] = outw;
}

// Pack Key9 key-side B fragments: pb[b][kb][ch][lane][dw]
__global__ void k_packB(const float* __restrict__ x, const float* __restrict__ invn,
                        unsigned* __restrict__ pb) {
  int idx = blockIdx.x * blockDim.x + threadIdx.x;
  if (idx >= BATCH * KBLK * NCH * 256) return;
  int dw = idx & 7, lane = (idx >> 3) & 31;
  int fragIdx = idx >> 8;
  int ch = fragIdx % NCH; int rest = fragIdx / NCH;
  int kb = rest % KBLK;  int b = rest / KBLK;
  int N = lane & 15, half = lane >> 4;
  int k = kb * 16 + N;
  unsigned outw = 0;
  for (int e = 0; e < 2; ++e) {
    int kk = (dw >> 2) * 16 + half * 8 + (dw & 3) * 2 + e;     // ISA 16-bit B 32x16 layout
    int j = ch * 32 + kk;
    int s = j >> 6, c = j & 63;
    int u = s / 3 - 1, t = s % 3 - 1;
    bool v; int g = gshift(k, u, t, v);
    float val = v ? x[((size_t)(b * 2 * CCH + CCH + c)) * LL + g] * invn[b * LL + g]
                  : 0.f;
    outw |= (unsigned)f2bf(val) << (16 * e);
  }
  pb[idx] = outw;
}

// Pack former^T B fragments for the output GEMM: pf[b][kp][nb][lane][dw]
__global__ void k_packF(const float* __restrict__ x, unsigned* __restrict__ pf) {
  int idx = blockIdx.x * blockDim.x + threadIdx.x;
  if (idx >= BATCH * KPAIR * 4 * 256) return;
  int dw = idx & 7, lane = (idx >> 3) & 31;
  int fragIdx = idx >> 8;
  int nb = fragIdx % 4; int rest = fragIdx / 4;
  int kp = rest % KPAIR; int b = rest / KPAIR;
  int N = lane & 15, half = lane >> 4;
  int cc = nb * 16 + N;
  unsigned outw = 0;
  for (int e = 0; e < 2; ++e) {
    int kk = (dw >> 2) * 16 + half * 8 + (dw & 3) * 2 + e;
    int k = kp * 32 + kk;
    float val = x[((size_t)(b * 2 * CCH + cc)) * LL + k];   // former channel
    outw |= (unsigned)f2bf(val) << (16 * e);
  }
  pf[idx] = outw;
}

// Copy former/latter straight into output channels 0..127
__global__ void k_copy(const float* __restrict__ x, float* __restrict__ out) {
  int idx = blockIdx.x * blockDim.x + threadIdx.x;
  if (idx >= BATCH * 2 * CCH * LL) return;
  int b = idx / (2 * CCH * LL);
  int r = idx % (2 * CCH * LL);
  out[(size_t)b * 3 * CCH * LL + r] = x[idx];
}

// ---------------- fused flash-attention main kernel ----------------

__global__ __launch_bounds__(256) void k_attn(const float* __restrict__ mask,
                                              const uint4* __restrict__ pa,
                                              const uint4* __restrict__ pb,
                                              const uint4* __restrict__ pf,
                                              float* __restrict__ out) {
  int wg = blockIdx.x;
  int b = wg >> 8, qb = wg & 255;
  int tid = threadIdx.x, wave = tid >> 5, lane = tid & 31;
  int laneN = lane & 15, half = lane >> 4;

  __shared__ uint4 sA[NCH * 64];              // 18 KB: query A fragments
  __shared__ unsigned short sP[8 * 512];      // 8 KB: per-wave P transpose buffer
  __shared__ float sO[1024];                  // 4 KB: 16x64 output accumulator
  __shared__ float sM[8][16], sS[8][16];
  __shared__ float sGM[16], sGS[16];

  const uint4* gA = pa + (size_t)(b * QBLK + qb) * NCH * 64;
  for (int i = tid; i < NCH * 64; i += 256) sA[i] = gA[i];
  for (int i = tid; i < 1024; i += 256) sO[i] = 0.f;
  __syncthreads();

  float mrun[8], srun[8];
  v8f O0 = {}, O1 = {}, O2 = {}, O3 = {};
#pragma unroll
  for (int i = 0; i < 8; ++i) { mrun[i] = -1e30f; srun[i] = 0.f; }

  const uint4* gBb = pb + (size_t)b * KBLK * NCH * 64;
  const uint4* gFb = pf + (size_t)b * KPAIR * 4 * 64;

  for (int kp = wave; kp < KPAIR; kp += 8) {
    // F fragments only depend on kp: issue first, hide under the score chain
    v16bf fB0 = ld_frag(&gFb[((size_t)kp * 4 + 0) * 64 + lane * 2]);
    v16bf fB1 = ld_frag(&gFb[((size_t)kp * 4 + 1) * 64 + lane * 2]);
    v16bf fB2 = ld_frag(&gFb[((size_t)kp * 4 + 2) * 64 + lane * 2]);
    v16bf fB3 = ld_frag(&gFb[((size_t)kp * 4 + 3) * 64 + lane * 2]);

    // warm L2 for the next key-pair block this wave will touch
    if (kp + 8 < KPAIR)
      __builtin_prefetch(&gBb[(size_t)(kp + 8) * 2 * NCH * 64 + lane * 2], 0, 1);

    const uint4* pb0 = &gBb[((size_t)(kp * 2 + 0) * NCH) * 64 + lane * 2];
    const uint4* pb1 = &gBb[((size_t)(kp * 2 + 1) * NCH) * 64 + lane * 2];

    // software-pipelined B fragment stream: load chunk ch+1 during chunk ch WMMAs
    v8f S0 = {}, S1 = {};
    v16bf nb0 = ld_frag(pb0);
    v16bf nb1 = ld_frag(pb1);
#pragma unroll
    for (int ch = 0; ch < NCH; ++ch) {
      v16bf b0 = nb0, b1 = nb1;
      if (ch + 1 < NCH) {
        nb0 = ld_frag(pb0 + (size_t)(ch + 1) * 64);
        nb1 = ld_frag(pb1 + (size_t)(ch + 1) * 64);
      }
      v16bf a = ld_frag(&sA[ch * 64 + lane * 2]);
      S0 = wmma_bf16(a, b0, S0);
      S1 = wmma_bf16(a, b1, S1);
    }

    int k0 = kp * 32 + laneN;
    bool f0 = mask[k0] > 0.5f;
    bool f1 = mask[k0 + 16] > 0.5f;

#pragma unroll
    for (int i = 0; i < 8; ++i) {
      float v0 = f0 ? -1e30f : S0[i];
      float v1 = f1 ? -1e30f : S1[i];
      float r = fmaxf(v0, v1);
#pragma unroll
      for (int sh = 1; sh <= 8; sh <<= 1) r = fmaxf(r, __shfl_xor(r, sh, 32));
      float nm = fmaxf(mrun[i], r);
      float al = __expf(mrun[i] - nm);
      float p0 = f0 ? 0.f : __expf(S0[i] - nm);
      float p1 = f1 ? 0.f : __expf(S1[i] - nm);
      float rs = p0 + p1;
#pragma unroll
      for (int sh = 1; sh <= 8; sh <<= 1) rs += __shfl_xor(rs, sh, 32);
      srun[i] = srun[i] * al + rs;
      mrun[i] = nm;
      int row = i + half * 8;
      sP[wave * 512 + row * 32 + laneN]      = f2bf(p0);   // D->row-major LDS
      sP[wave * 512 + row * 32 + 16 + laneN] = f2bf(p1);
      O0[i] *= al; O1[i] *= al; O2[i] *= al; O3[i] *= al;
    }
    asm volatile("s_wait_dscnt 0" ::: "memory");   // P visible to own wave

    // reload P as an A fragment (16x32 bf16, ISA layout)
    const uint4* pw = (const uint4*)(sP + wave * 512);
    uint4 plo = pw[laneN * 4 + half];
    uint4 phi = pw[laneN * 4 + 2 + half];
    Frag2 fr; fr.lo = plo; fr.hi = phi;
    v16bf aP = __builtin_bit_cast(v16bf, fr);

    O0 = wmma_bf16(aP, fB0, O0);
    O1 = wmma_bf16(aP, fB1, O1);
    O2 = wmma_bf16(aP, fB2, O2);
    O3 = wmma_bf16(aP, fB3, O3);
  }

  // ---- inter-wave log-sum-exp merge ----
  if (laneN == 0) {
#pragma unroll
    for (int i = 0; i < 8; ++i) {
      sM[wave][i + half * 8] = mrun[i];
      sS[wave][i + half * 8] = srun[i];
    }
  }
  __syncthreads();
  if (tid < 16) {
    float gm = -1e30f;
    for (int w = 0; w < 8; ++w) gm = fmaxf(gm, sM[w][tid]);
    float gs = 0.f;
    for (int w = 0; w < 8; ++w) gs += sS[w][tid] * __expf(sM[w][tid] - gm);
    sGM[tid] = gm; sGS[tid] = gs;
  }
  __syncthreads();
#pragma unroll
  for (int i = 0; i < 8; ++i) {
    int row = i + half * 8;
    float sc = __expf(mrun[i] - sGM[row]);
    atomicAdd(&sO[row * 64 + laneN],      O0[i] * sc);
    atomicAdd(&sO[row * 64 + 16 + laneN], O1[i] * sc);
    atomicAdd(&sO[row * 64 + 32 + laneN], O2[i] * sc);
    atomicAdd(&sO[row * 64 + 48 + laneN], O3[i] * sc);
  }
  __syncthreads();

  for (int i = tid; i < 1024; i += 256) {
    int row = i >> 6, col = i & 63;
    int l = qb * 16 + row;
    float fl = (mask[l] > 0.5f) ? 1.f : 0.f;
    float gs = sGS[row];
    float val = (gs > 0.f) ? (sO[i] / gs) : 0.f;
    out[((size_t)b * 3 * CCH + 2 * CCH + col) * LL + l] = val * fl;
  }
}

// ---------------- launcher ----------------

extern "C" void kernel_launch(void* const* d_in, const int* in_sizes, int n_in,
                              void* d_out, int out_size, void* d_ws, size_t ws_size,
                              hipStream_t stream) {
  (void)in_sizes; (void)n_in; (void)out_size; (void)ws_size;
  const float* x    = (const float*)d_in[0];
  const float* mask = (const float*)d_in[1];
  float* out = (float*)d_out;

  char* ws = (char*)d_ws;
  size_t offInv = 0;
  size_t offA = offInv + (size_t)BATCH * LL * sizeof(float);               // 32 KB
  size_t offB = offA + (size_t)BATCH * QBLK * NCH * 1024;                  // 9.4 MB
  size_t offF = offB + (size_t)BATCH * KBLK * NCH * 1024;                  // 9.4 MB
  float*    invn = (float*)(ws + offInv);
  unsigned* pa   = (unsigned*)(ws + offA);
  unsigned* pb   = (unsigned*)(ws + offB);
  unsigned* pf   = (unsigned*)(ws + offF);                                 // 1 MB

  {
    int n = BATCH * LL;
    k_invnorm<<<(n + 255) / 256, 256, 0, stream>>>(x, invn);
  }
  {
    int n = BATCH * QBLK * NCH * 256;
    k_packA<<<(n + 255) / 256, 256, 0, stream>>>(x, pa);
  }
  {
    int n = BATCH * KBLK * NCH * 256;
    k_packB<<<(n + 255) / 256, 256, 0, stream>>>(x, invn, pb);
  }
  {
    int n = BATCH * KPAIR * 4 * 256;
    k_packF<<<(n + 255) / 256, 256, 0, stream>>>(x, pf);
  }
  {
    int n = BATCH * 2 * CCH * LL;
    k_copy<<<(n + 255) / 256, 256, 0, stream>>>(x, out);
  }
  k_attn<<<BATCH * QBLK, 256, 0, stream>>>(mask, (const uint4*)pa,
                                           (const uint4*)pb, (const uint4*)pf, out);
}
